// WTM_A_16879221473457
// MI455X (gfx1250) — compile-verified, matrix operand order checked by reading, not run on hardware
//
#include <hip/hip_runtime.h>
#include <math.h>

typedef __bf16 bf16_t;
typedef __attribute__((ext_vector_type(16))) __bf16 v16bf;
typedef __attribute__((ext_vector_type(8)))  __bf16 v8bf;
typedef __attribute__((ext_vector_type(8)))  float  v8f;

union Frag { v16bf v; v8bf h[2]; };

#define DEV __device__ __forceinline__

static constexpr int HH = 48, WWD = 36, NPIX = HH * WWD;       // 1728 tokens
static constexpr float BN_SCL = 0.99999500003749978f;          // 1/sqrt(1+1e-5)

DEV unsigned short f2bf(float f) {
  unsigned int u = __float_as_uint(f);
  return (unsigned short)((u + 0x7FFFu + ((u >> 16) & 1u)) >> 16);
}

// ---------------------------------------------------------------------------
// Generic WMMA GEMM:  C[m,n] = sum_k A[m,k]*B[n,k]
// A: MxK bf16 row-major, B: NxK bf16 row-major (weights as stored [O,C]).
// One wave -> 16 x (16*NT) tile. 8 waves / block.
// Fragment layout per CDNA5 ISA 7.12.2:
//   A: lane l holds row m=l%15.. (l&15), half=l>>4; elem e -> K = (e/8)*16 + half*8 + e%8
//   B: lane l holds col n=l&15 with identical K ordering -> two b128 loads/lane.
//   C/D: v8f, elem r -> m = r + 8*half, n = l&15.
// Epilogue modes:
//   0: BN+ReLU -> optional f32 [m,ld32]+bf16 [m,ldbf]
//   1: +bias   -> f32
//   2: +bias, GELU(erf) -> bf16
//   3: resid[m,N] += scale[n]*(acc+bias[n])   (layerscale residual)
//   4: BN+ReLU -> f32 transposed NCHW: out32[n*ld32 + m]
// ---------------------------------------------------------------------------
template <int NT>
__global__ void __launch_bounds__(256)
wmma_gemm(const unsigned short* __restrict__ Aq,
          const unsigned short* __restrict__ Bq,
          int M, int N, int K,
          const float* __restrict__ scale,
          const float* __restrict__ shift,
          float* __restrict__ out32, int ld32,
          unsigned short* __restrict__ outbf, int ldbf,
          float* __restrict__ resid,
          int mode)
{
  const bf16_t* A = reinterpret_cast<const bf16_t*>(Aq);
  const bf16_t* B = reinterpret_cast<const bf16_t*>(Bq);
  const int wave = threadIdx.x >> 5;
  const int lane = threadIdx.x & 31;
  const int hl = lane >> 4;   // lane half (K-group select)
  const int lm = lane & 15;   // row (A) / col (B) within tile

  const int tilesN = N / (16 * NT);
  const int tiles = (M / 16) * tilesN;
  const int tid = blockIdx.x * 8 + wave;       // wave-uniform
  if (tid >= tiles) return;                    // whole-wave predicate: EXEC stays full
  const int tM = tid / tilesN;
  const int tN = tid % tilesN;

  const bf16_t* Ap = A + (size_t)(tM * 16 + lm) * K + hl * 8;
  const bf16_t* Bp[NT];
#pragma unroll
  for (int t = 0; t < NT; ++t)
    Bp[t] = B + (size_t)((tN * NT + t) * 16 + lm) * K + hl * 8;

  v8f acc[NT] = {};
  for (int k0 = 0; k0 < K; k0 += 32) {
    Frag a;
    a.h[0] = *(const v8bf*)(Ap + k0);
    a.h[1] = *(const v8bf*)(Ap + k0 + 16);
#pragma unroll
    for (int t = 0; t < NT; ++t) {
      Frag bm;
      bm.h[0] = *(const v8bf*)(Bp[t] + k0);
      bm.h[1] = *(const v8bf*)(Bp[t] + k0 + 16);
      acc[t] = __builtin_amdgcn_wmma_f32_16x16x32_bf16(
          false, a.v, false, bm.v, (short)0, acc[t], false, false);
    }
  }

#pragma unroll
  for (int t = 0; t < NT; ++t) {
    const int n = (tN * NT + t) * 16 + lm;
    const float sc = scale ? scale[n] : 1.f;
    const float sf = shift ? shift[n] : 0.f;
#pragma unroll
    for (int r = 0; r < 8; ++r) {
      const int m = tM * 16 + hl * 8 + r;
      float v = acc[t][r];
      if (mode == 0) {
        v = fmaxf(v * (sc * BN_SCL) + sf, 0.f);
        if (out32) out32[(size_t)m * ld32 + n] = v;
        if (outbf) outbf[(size_t)m * ldbf + n] = f2bf(v);
      } else if (mode == 1) {
        out32[(size_t)m * ld32 + n] = v + sf;
      } else if (mode == 2) {
        const float u = v + sf;
        outbf[(size_t)m * ldbf + n] = f2bf(0.5f * u * (1.f + erff(u * 0.70710678118654752f)));
      } else if (mode == 3) {
        const size_t idx = (size_t)m * N + n;
        resid[idx] += sc * (v + sf);
      } else { // mode 4: BN+ReLU, NCHW transposed store
        v = fmaxf(v * (sc * BN_SCL) + sf, 0.f);
        out32[(size_t)n * ld32 + m] = v;
      }
    }
  }
}

// -------------------------- helpers / small kernels ------------------------
__global__ void f32_to_bf16_kernel(const float* __restrict__ s,
                                   unsigned short* __restrict__ d, int n) {
  int t = blockIdx.x * blockDim.x + threadIdx.x;
  if (t < n) d[t] = f2bf(s[t]);
}

DEV float bilin(const float* __restrict__ src, int c, int h, int w, int y, int x) {
  float yc = (h > 1) ? y * ((float)(h - 1) / (float)(HH - 1)) : 0.f;
  float xc = (w > 1) ? x * ((float)(w - 1) / (float)(WWD - 1)) : 0.f;
  int y0 = (int)floorf(yc), x0 = (int)floorf(xc);
  int y1 = min(y0 + 1, h - 1), x1 = min(x0 + 1, w - 1);
  float wy = yc - (float)y0, wx = xc - (float)x0;
  const float* p = src + (size_t)c * h * w;
  float g00 = p[y0 * w + x0], g01 = p[y0 * w + x1];
  float g10 = p[y1 * w + x0], g11 = p[y1 * w + x1];
  return g00 * (1.f - wy) * (1.f - wx) + g01 * (1.f - wy) * wx +
         g10 * wy * (1.f - wx) + g11 * wy * wx;
}

// concat[ inputs0 | up(i1) | up(i2) | up(i3) ] -> bf16 HWC [1728,1440]
__global__ void build_xcat_kernel(const float* __restrict__ i0, const float* __restrict__ i1,
                                  const float* __restrict__ i2, const float* __restrict__ i3,
                                  unsigned short* __restrict__ xcat) {
  int t = blockIdx.x * blockDim.x + threadIdx.x;
  if (t >= NPIX * 1440) return;
  int n = t / 1440, c = t % 1440;
  int y = n / WWD, x = n % WWD;
  float v;
  if (c < 96)        v = i0[(size_t)c * NPIX + n];
  else if (c < 288)  v = bilin(i1, c - 96, 24, 18, y, x);
  else if (c < 672)  v = bilin(i2, c - 288, 12, 9, y, x);
  else               v = bilin(i3, c - 672, 6, 5, y, x);
  xcat[(size_t)n * 1440 + c] = f2bf(v);
}

// LayerNorm over C=256, one wave per row, write bf16
__global__ void __launch_bounds__(256)
ln_kernel(const float* __restrict__ x, const float* __restrict__ g,
          const float* __restrict__ b, unsigned short* __restrict__ o) {
  int wave = threadIdx.x >> 5;
  int lane = threadIdx.x & 31;
  int row = blockIdx.x * 8 + wave;
  const float* xr = x + (size_t)row * 256;
  float v[8], s = 0.f, q = 0.f;
#pragma unroll
  for (int j = 0; j < 8; ++j) { v[j] = xr[lane + j * 32]; s += v[j]; q += v[j] * v[j]; }
#pragma unroll
  for (int m = 16; m >= 1; m >>= 1) { s += __shfl_xor(s, m, 32); q += __shfl_xor(q, m, 32); }
  float mean = s * (1.f / 256.f);
  float var = q * (1.f / 256.f) - mean * mean;
  float inv = rsqrtf(var + 1e-5f);
  unsigned short* orow = o + (size_t)row * 256;
#pragma unroll
  for (int j = 0; j < 8; ++j) {
    int c = lane + j * 32;
    orow[c] = f2bf((v[j] - mean) * inv * g[c] + b[c]);
  }
}

DEV int wstart(int i, int L, int d) {
  if (d <= 1) {
    int s = i - 3; if (s < 0) s = 0;
    if (i + 3 >= L) s = L - 7;
    return s;
  }
  int ni = i - 3 * d;
  if (ni < 0) return i % d;
  if (i + 3 * d >= L) {
    int im = i % d;
    int a = (L / d) * d;
    int b = L - a;
    if (im < b) return L - b + im - 6 * d;
    return a + im - 7 * d;
  }
  return ni;
}
DEV int fdiv(int a, int d) { return (a >= 0) ? a / d : -((-a + d - 1) / d); }

// Neighborhood attention, one thread per (head, pixel). qkv fp32 [1728,768].
__global__ void __launch_bounds__(128)
nat_attn_kernel(const float* __restrict__ qkv, const float* __restrict__ rpb,
                unsigned short* __restrict__ out, int dil) {
  int t = blockIdx.x * blockDim.x + threadIdx.x;
  if (t >= NPIX * 8) return;
  int head = t / NPIX;
  int n = t % NPIX;
  int y = n / WWD, x = n % WWD;
  int sh = wstart(y, HH, dil);
  int sw = wstart(x, WWD, dil);
  const float* qb = qkv + (size_t)n * 768 + head * 32;
  float q[32];
#pragma unroll
  for (int c = 0; c < 32; ++c) q[c] = qb[c] * 0.17677669529663687f; // 32^-0.5
  float logit[49];
  float mx = -1e30f;
#pragma unroll
  for (int jh = 0; jh < 7; ++jh) {
    int ih = sh + jh * dil;
    int bh = fdiv(ih - y, dil) + 6;
#pragma unroll
    for (int jw = 0; jw < 7; ++jw) {
      int iw = sw + jw * dil;
      int bw = fdiv(iw - x, dil) + 6;
      const float* kb = qkv + (size_t)(ih * WWD + iw) * 768 + 256 + head * 32;
      float s = 0.f;
#pragma unroll
      for (int c = 0; c < 32; ++c) s += q[c] * kb[c];
      s += rpb[(head * 13 + bh) * 13 + bw];
      logit[jh * 7 + jw] = s;
      mx = fmaxf(mx, s);
    }
  }
  float sum = 0.f;
#pragma unroll
  for (int j = 0; j < 49; ++j) { logit[j] = __expf(logit[j] - mx); sum += logit[j]; }
  float inv = 1.f / sum;
  float acc[32];
#pragma unroll
  for (int c = 0; c < 32; ++c) acc[c] = 0.f;
#pragma unroll
  for (int jh = 0; jh < 7; ++jh) {
    int ih = sh + jh * dil;
#pragma unroll
    for (int jw = 0; jw < 7; ++jw) {
      int iw = sw + jw * dil;
      const float* vb = qkv + (size_t)(ih * WWD + iw) * 768 + 512 + head * 32;
      float w = logit[jh * 7 + jw] * inv;
#pragma unroll
      for (int c = 0; c < 32; ++c) acc[c] += w * vb[c];
    }
  }
#pragma unroll
  for (int c = 0; c < 32; ++c)
    out[(size_t)n * 256 + head * 32 + c] = f2bf(acc[c]);
}

__global__ void colmean_kernel(const float* __restrict__ x, float* __restrict__ mean) {
  int c = blockIdx.x * blockDim.x + threadIdx.x;
  if (c >= 256) return;
  float s = 0.f;
  for (int n = 0; n < NPIX; ++n) s += x[(size_t)n * 256 + c];
  mean[c] = s * (1.f / (float)NPIX);
}

__global__ void gap_fc_kernel(const float* __restrict__ mean, const float* __restrict__ w,
                              const float* __restrict__ g, const float* __restrict__ b,
                              float* __restrict__ o) {
  int oc = blockIdx.x * blockDim.x + threadIdx.x;
  if (oc >= 256) return;
  float s = 0.f;
  for (int c = 0; c < 256; ++c) s += w[oc * 256 + c] * mean[c];
  o[oc] = fmaxf(s * (g[oc] * BN_SCL) + b[oc], 0.f);
}

__global__ void bcast_x5_kernel(const float* __restrict__ x5, unsigned short* __restrict__ cat) {
  int t = blockIdx.x * blockDim.x + threadIdx.x;
  if (t >= NPIX * 256) return;
  int n = t / 256, c = t % 256;
  cat[(size_t)n * 1280 + 1024 + c] = f2bf(x5[c]);
}

// fp32 [rows,cols] (ld=lds) -> bf16 slice (ld=ldd); dst already offset
__global__ void copy_slice_kernel(const float* __restrict__ src, unsigned short* __restrict__ dst,
                                  int rows, int cols, int lds, int ldd) {
  int t = blockIdx.x * blockDim.x + threadIdx.x;
  if (t >= rows * cols) return;
  int r = t / cols, c = t % cols;
  dst[(size_t)r * ldd + c] = f2bf(src[(size_t)r * lds + c]);
}

// NCHW fp32 [C,1728] -> HWC bf16 [1728,C] for C=256
__global__ void chw2hwc_kernel(const float* __restrict__ src, unsigned short* __restrict__ dst) {
  int t = blockIdx.x * blockDim.x + threadIdx.x;
  if (t >= NPIX * 256) return;
  int c = t / NPIX, n = t % NPIX;
  dst[(size_t)n * 256 + c] = f2bf(src[(size_t)c * NPIX + n]);
}

// im2col for 3x3 same-pad conv; in bf16 HWC [1728,C] -> [1728, C*9], k = c*9+ky*3+kx
__global__ void im2col3_kernel(const unsigned short* __restrict__ in,
                               unsigned short* __restrict__ out, int C) {
  int t = blockIdx.x * blockDim.x + threadIdx.x;
  if (t >= NPIX * C) return;
  int n = t / C, c = t % C;
  int y = n / WWD, x = n % WWD;
  size_t ob = (size_t)n * C * 9 + (size_t)c * 9;
#pragma unroll
  for (int ky = 0; ky < 3; ++ky) {
    int yy = y + ky - 1;
#pragma unroll
    for (int kx = 0; kx < 3; ++kx) {
      int xx = x + kx - 1;
      unsigned short v = 0;
      if (yy >= 0 && yy < HH && xx >= 0 && xx < WWD)
        v = in[(size_t)(yy * WWD + xx) * C + c];
      out[ob + ky * 3 + kx] = v;
    }
  }
}

// ---------------------------------------------------------------------------
extern "C" void kernel_launch(void* const* d_in, const int* in_sizes, int n_in,
                              void* d_out, int out_size, void* d_ws, size_t ws_size,
                              hipStream_t stream) {
  (void)in_sizes; (void)n_in; (void)out_size; (void)ws_size;
  auto F = [&](int i) { return (const float*)d_in[i]; };
  // input leaf order (recursive dict insertion order):
  // 0..3 inputs0..3, 4 low_level_feat, 5 conv1_w, 6 bn1.g, 7 bn1.b,
  // per NAT layer li (base 8+15*li): ln1_g,ln1_b,qkv_w,qkv_b,rpb,proj_w,proj_b,
  //   gamma1,gamma2,ln2_g,ln2_b,fc1_w,fc1_b,fc2_w,fc2_b
  // 128 gap_w, 129/130 gap_bn, 131 conv2_w, 132/133 bn2, 134 low_w,
  // 135/136 bn_low, 137 lc1_w, 138/139 lc_bn1, 140 lc2_w, 141/142 lc_bn2

  char* base = (char*)d_ws;
  size_t off = 0;
  auto alloc = [&](size_t bytes) -> void* {
    void* p = base + off;
    off = (off + bytes + 255) & ~(size_t)255;
    return p;
  };
  unsigned short* wcat = (unsigned short*)alloc((size_t)NPIX * 1440 * 2);
  float* y             = (float*)alloc((size_t)NPIX * 256 * 4);
  unsigned short* lnb  = (unsigned short*)alloc((size_t)NPIX * 256 * 2);
  float* qkvact        = (float*)alloc((size_t)NPIX * 768 * 4);
  unsigned short* attn = (unsigned short*)alloc((size_t)NPIX * 256 * 2);
  unsigned short* hid  = (unsigned short*)alloc((size_t)NPIX * 1024 * 2);
  unsigned short* cat  = (unsigned short*)alloc((size_t)NPIX * 1280 * 2);
  float* meanb         = (float*)alloc(256 * 4);
  float* gapv          = (float*)alloc(256 * 4);
  unsigned short* lowh = (unsigned short*)alloc((size_t)NPIX * 256 * 2);
  unsigned short* cat2 = (unsigned short*)alloc((size_t)NPIX * 288 * 2);
  unsigned short* col1 = (unsigned short*)alloc((size_t)NPIX * 2592 * 2);
  unsigned short* tbuf = (unsigned short*)alloc((size_t)NPIX * 256 * 2);
  unsigned short* col2 = (unsigned short*)alloc((size_t)NPIX * 2304 * 2);
  unsigned short* wcv1 = (unsigned short*)alloc((size_t)256 * 1440 * 2);
  unsigned short *wqkv[8], *wproj[8], *wfc1[8], *wfc2[8];
  for (int li = 0; li < 8; ++li) {
    wqkv[li]  = (unsigned short*)alloc((size_t)768 * 256 * 2);
    wproj[li] = (unsigned short*)alloc((size_t)256 * 256 * 2);
    wfc1[li]  = (unsigned short*)alloc((size_t)1024 * 256 * 2);
    wfc2[li]  = (unsigned short*)alloc((size_t)256 * 1024 * 2);
  }
  unsigned short* wcv2 = (unsigned short*)alloc((size_t)256 * 1280 * 2);
  unsigned short* wlow = (unsigned short*)alloc((size_t)32 * 256 * 2);
  unsigned short* wlc1 = (unsigned short*)alloc((size_t)256 * 2592 * 2);
  unsigned short* wlc2 = (unsigned short*)alloc((size_t)256 * 2304 * 2);

  auto cvt = [&](const float* s, unsigned short* d, int n) {
    f32_to_bf16_kernel<<<(n + 255) / 256, 256, 0, stream>>>(s, d, n);
  };
  auto gemm = [&](int NT, const unsigned short* A, const unsigned short* B,
                  int M, int N, int K, const float* sc, const float* sh,
                  float* o32, int ld32, unsigned short* obf, int ldbf,
                  float* res, int mode) {
    int tiles = (M / 16) * (N / (16 * NT));
    int blocks = (tiles + 7) / 8;
    if (NT == 4)
      wmma_gemm<4><<<blocks, 256, 0, stream>>>(A, B, M, N, K, sc, sh, o32, ld32, obf, ldbf, res, mode);
    else
      wmma_gemm<2><<<blocks, 256, 0, stream>>>(A, B, M, N, K, sc, sh, o32, ld32, obf, ldbf, res, mode);
  };

  // --- weight conversion (fp32 -> bf16) ---
  cvt(F(5), wcv1, 256 * 1440);
  for (int li = 0; li < 8; ++li) {
    int pb = 8 + 15 * li;
    cvt(F(pb + 2),  wqkv[li], 768 * 256);
    cvt(F(pb + 5),  wproj[li], 256 * 256);
    cvt(F(pb + 11), wfc1[li], 1024 * 256);
    cvt(F(pb + 13), wfc2[li], 256 * 1024);
  }
  cvt(F(131), wcv2, 256 * 1280);
  cvt(F(134), wlow, 32 * 256);
  cvt(F(137), wlc1, 256 * 2592);
  cvt(F(140), wlc2, 256 * 2304);

  // --- stem: bilinear concat + conv1(1440->256)+BN+ReLU ---
  build_xcat_kernel<<<(NPIX * 1440 + 255) / 256, 256, 0, stream>>>(F(0), F(1), F(2), F(3), wcat);
  gemm(4, wcat, wcv1, NPIX, 256, 1440, F(6), F(7), y, 256, nullptr, 0, nullptr, 0);

  // --- GAP branch (uses pre-NAT features) ---
  colmean_kernel<<<1, 256, 0, stream>>>(y, meanb);
  gap_fc_kernel<<<1, 256, 0, stream>>>(meanb, F(128), F(129), F(130), gapv);

  // --- 8 NAT layers ---
  static const int DILS_[8] = {1, 2, 1, 3, 1, 4, 1, 5};
  for (int li = 0; li < 8; ++li) {
    int pb = 8 + 15 * li;
    ln_kernel<<<NPIX / 8, 256, 0, stream>>>(y, F(pb + 0), F(pb + 1), lnb);
    gemm(4, lnb, wqkv[li], NPIX, 768, 256, nullptr, F(pb + 3), qkvact, 768, nullptr, 0, nullptr, 1);
    nat_attn_kernel<<<(NPIX * 8 + 127) / 128, 128, 0, stream>>>(qkvact, F(pb + 4), attn, DILS_[li]);
    gemm(4, attn, wproj[li], NPIX, 256, 256, F(pb + 7), F(pb + 6), nullptr, 0, nullptr, 0, y, 3);
    ln_kernel<<<NPIX / 8, 256, 0, stream>>>(y, F(pb + 9), F(pb + 10), lnb);
    gemm(4, lnb, wfc1[li], NPIX, 1024, 256, nullptr, F(pb + 12), nullptr, 0, hid, 1024, nullptr, 2);
    gemm(4, hid, wfc2[li], NPIX, 256, 1024, F(pb + 8), F(pb + 14), nullptr, 0, nullptr, 0, y, 3);
    if (li & 1)
      copy_slice_kernel<<<(NPIX * 256 + 255) / 256, 256, 0, stream>>>(
          y, cat + (li / 2) * 256, NPIX, 256, 256, 1280);
  }

  // --- concat [x1..x4, x5] -> conv2(1280->256)+BN+ReLU -> cat2[:, :256] ---
  bcast_x5_kernel<<<(NPIX * 256 + 255) / 256, 256, 0, stream>>>(gapv, cat);
  gemm(4, cat, wcv2, NPIX, 256, 1280, F(132), F(133), nullptr, 0, cat2, 288, nullptr, 0);

  // --- low-level branch: conv1x1(256->32)+BN+ReLU -> cat2[:, 256:288] ---
  chw2hwc_kernel<<<(NPIX * 256 + 255) / 256, 256, 0, stream>>>(F(4), lowh);
  gemm(2, lowh, wlow, NPIX, 32, 256, F(135), F(136), nullptr, 0, cat2 + 256, 288, nullptr, 0);

  // --- 3x3 conv (288->256) + BN + ReLU via im2col GEMM ---
  im2col3_kernel<<<(NPIX * 288 + 255) / 256, 256, 0, stream>>>(cat2, col1, 288);
  gemm(4, col1, wlc1, NPIX, 256, 2592, F(138), F(139), nullptr, 0, tbuf, 256, nullptr, 0);

  // --- 3x3 conv (256->256) + BN + ReLU, store fp32 NCHW to d_out ---
  im2col3_kernel<<<(NPIX * 256 + 255) / 256, 256, 0, stream>>>(tbuf, col2, 256);
  gemm(4, col2, wlc2, NPIX, 256, 2304, F(141), F(142), (float*)d_out, NPIX, nullptr, 0, nullptr, 4);
}
